// DecoderLSTM_36584531427908
// MI455X (gfx1250) — compile-verified, matrix operand order checked by reading, not run on hardware
//
#include <hip/hip_runtime.h>
#include <hip/hip_bf16.h>

typedef __attribute__((ext_vector_type(16))) __bf16 v16bf;
typedef __attribute__((ext_vector_type(8)))  float  v8f;

#define NWG 32
#define BATCH 256
#define HDIM 256

__device__ __forceinline__ float sigm(float x)     { return 1.f / (1.f + __expf(-x)); }
__device__ __forceinline__ float tanhfast(float x) { return 1.f - 2.f / (__expf(2.f * x) + 1.f); }

// ---------------- barrier ----------------
__global__ void bar_init(unsigned* bar) { bar[0] = 0u; bar[1] = 0u; }

__device__ __forceinline__ void grid_barrier(unsigned* bar) {
  __threadfence();
  __syncthreads();
  if (threadIdx.x == 0) {
    unsigned g = __atomic_load_n(bar + 1, __ATOMIC_ACQUIRE);
    if (__atomic_fetch_add(bar, 1u, __ATOMIC_ACQ_REL) == NWG - 1) {
      __atomic_store_n(bar, 0u, __ATOMIC_RELAXED);
      __atomic_fetch_add(bar + 1, 1u, __ATOMIC_RELEASE);
    } else {
      while (__atomic_load_n(bar + 1, __ATOMIC_ACQUIRE) == g)
        __builtin_amdgcn_s_sleep(1);
    }
  }
  __syncthreads();
  __threadfence();
}

// ---------------- weight pack: [Wih;Whh]^T -> fragment-native bf16 ----------------
// For K-block kb, column n: 32 bf16 stored as two 16-elem chunks (one per lane half)
// matching the CDNA5 16-bit B-matrix K interleave, so a lane loads one 32B v16bf.
__global__ void pack_w(const float* __restrict__ Wih, const float* __restrict__ Whh,
                       __hip_bfloat16* __restrict__ dst, int Kx, int DinReal) {
  int K_total = Kx + HDIM;
  int idx = blockIdx.x * blockDim.x + threadIdx.x;
  if (idx >= K_total * 1024) return;
  int k = idx >> 10;
  int n = idx & 1023;
  float val;
  if (k < Kx) val = (k < DinReal) ? Wih[(size_t)n * DinReal + k] : 0.f;
  else        val = Whh[(size_t)n * HDIM + (k - Kx)];
  int kb = k >> 5, kp = k & 31;
  int half = (kp >> 3) & 1;                       // K interleave per lane-half
  int pos  = (kp & 7) + ((kp & 16) ? 8 : 0);
  dst[(((size_t)(kb << 10) + n) << 5) + half * 16 + pos] = __float2bfloat16(val);
}

// ---------------- stage-0 input gather: bb = x1[:, {0,72,144,216}, 0:2] pad->32 ----
__global__ void gather0(const float* __restrict__ x1, __hip_bfloat16* __restrict__ dst) {
  int idx = blockIdx.x * blockDim.x + threadIdx.x;
  if (idx >= BATCH * 4 * 32) return;
  int c = idx & 31, j = (idx >> 5) & 3, b = idx >> 7;
  float v = (c < 2) ? x1[((size_t)b * 288 + j * 72) * 2 + c] : 0.f;
  dst[idx] = __float2bfloat16(v);
}

// ---------------- time-repeat upsample (f32 stage out -> bf16 next-stage in) -------
__global__ void repeat_bf16(const float* __restrict__ prev, __hip_bfloat16* __restrict__ dst,
                            int T, int rep, int Tp) {
  size_t idx = (size_t)blockIdx.x * blockDim.x + threadIdx.x;
  size_t total = (size_t)BATCH * T * 512;
  if (idx >= total) return;
  int c = (int)(idx & 511);
  int t = (int)((idx >> 9) % T);
  int b = (int)(idx / ((size_t)T * 512));
  dst[idx] = __float2bfloat16(prev[((size_t)b * Tp + t / rep) * 512 + c]);
}

// ---------------- stage head: out[b,s] = dot(out_f32[b,s,:512], w) + bias ----------
__global__ void head_k(const float* __restrict__ o, const float* __restrict__ w,
                       const float* __restrict__ bias, float* __restrict__ dst, int T) {
  int idx = blockIdx.x * blockDim.x + threadIdx.x;
  if (idx >= BATCH * T) return;
  const float* row = o + (size_t)idx * 512;
  float acc = 0.f;
#pragma unroll 8
  for (int k = 0; k < 512; ++k) acc += row[k] * w[k];
  dst[idx] = acc + bias[0];
}

// A fragment from a byte pointer: two contiguous 16B runs at +0 and +32 bytes
// (CDNA5 16-bit A-matrix K interleave, half already folded into the pointer).
__device__ __forceinline__ v16bf load_afrag(const char* p) {
  union { v16bf v; uint4 q[2]; } A;
  A.q[0] = *reinterpret_cast<const uint4*>(p);
  A.q[1] = *reinterpret_cast<const uint4*>(p + 32);
  return A.v;
}

// B fragment f (0..7) of the block at byte base bp: j = f&1, g = f>>1.
__device__ __forceinline__ v16bf load_bfrag(const char* bp, int f) {
  return *reinterpret_cast<const v16bf*>(bp + (f & 1) * 1024 + (f >> 1) * 16384);
}

// One K-block: 8 WMMAs through the rolling 2-deep B pipeline (3 frags in flight);
// the lookahead loads cross into the next block (the final block reads the pad).
#define WMMA_BLOCK(BP)                                                      \
  {                                                                         \
    _Pragma("unroll")                                                       \
    for (int f = 0; f < 8; ++f) {                                           \
      v16bf bn = (f < 6) ? load_bfrag((BP), f + 2)                          \
                         : load_bfrag((BP) + 65536, f - 6);                 \
      acc[f & 1][f >> 1] = __builtin_amdgcn_wmma_f32_16x16x32_bf16(         \
          false, a, false, bcur, (short)0, acc[f & 1][f >> 1],              \
          false, false);                                                    \
      bcur = bnx1; bnx1 = bn;                                               \
    }                                                                       \
  }

// ---------------- persistent bidirectional LSTM layer ------------------------------
// grid = 32 WGs x 256 thr = 256 waves = 2 dirs x 16 row-tiles x 8 wave-col-groups,
// each wave owns 2 col-groups (2 jobs), each job = matched i/f/g/o 16x16 tiles.
// The weight base gets an opaque (asm) zero OFFSET each step - LICM cannot hoist the
// weight loads across t, while pointer provenance stays global (global_load, not
// flat_load). A and B streams advance via rolling byte pointers; boundary blocks are
// peeled so every block body is branch-free with immediate-offset loads only.
template <int KBX>
__global__ void __launch_bounds__(256) lstm_layer(
    const __hip_bfloat16* __restrict__ in, int T,
    const __hip_bfloat16* __restrict__ WpF, const __hip_bfloat16* __restrict__ WpB,
    const float* __restrict__ biasF, const float* __restrict__ biasB,
    const float* __restrict__ h0F, const float* __restrict__ c0F,
    const float* __restrict__ h0B, const float* __restrict__ c0B,
    float* __restrict__ outF32, __hip_bfloat16* __restrict__ outBf, int lastLayer,
    const float* __restrict__ resid, int rep, int Tp,
    __hip_bfloat16* __restrict__ hstate,  // [2 dirs][2 bufs][256*256] bf16
    unsigned* __restrict__ bar) {
  constexpr int Kx  = KBX * 32;
  constexpr int KBH = HDIM / 32;  // 8 h-region K-blocks
  const int lane = threadIdx.x & 31;
  const int wave = blockIdx.x * 8 + (threadIdx.x >> 5);   // 0..255
  const int d    = wave >> 7;                             // direction
  const int r    = (wave >> 3) & 15;                      // row tile
  const int cgb  = (wave & 7) * 2;                        // first col-group of 2
  const int m0   = r * 16;
  const int half = lane >> 4;
  const int nl15 = lane & 15;

  const __hip_bfloat16* Wp   = d ? WpB : WpF;
  const float* bias          = d ? biasB : biasF;
  const float* h0            = d ? h0B : h0F;
  const float* c0            = d ? c0B : c0F;

  // ---- init: c into registers, h0 into the bf16 state buffer
  float creg[2][8];
#pragma unroll
  for (int j = 0; j < 2; ++j) {
    int col = (cgb + j) * 16 + nl15;
#pragma unroll
    for (int v = 0; v < 8; ++v) {
      int row = m0 + v + 8 * half;
      size_t sidx = (size_t)row * HDIM + col;
      creg[j][v] = c0[sidx];
      hstate[((size_t)d * 2 + 0) * 65536 + sidx] = __float2bfloat16(h0[sidx]);
    }
  }
  grid_barrier(bar);

  // per-lane bias values (replicated across the 8 accumulator rows)
  float bvals[2][4];
#pragma unroll
  for (int j = 0; j < 2; ++j)
#pragma unroll
    for (int g = 0; g < 4; ++g)
      bvals[j][g] = bias[g * HDIM + (cgb + j) * 16 + nl15];

  // single weight base (j=0); j=1 is +1024 B, gate g is +g*16384 B, block is +65536 B
  const char* bb0 = (const char*)Wp +
                    ((size_t)(cgb * 16 + nl15) * 32 + half * 16) * sizeof(__hip_bfloat16);

  const int rowA = m0 + nl15;  // A-fragment row for this lane

  for (int t = 0; t < T; ++t) {
    const int tt = d ? (T - 1 - t) : t;
    const __hip_bfloat16* hcur = hstate + ((size_t)d * 2 + (t & 1)) * 65536;

    v8f acc[2][4];
#pragma unroll
    for (int j = 0; j < 2; ++j)
#pragma unroll
      for (int g = 0; g < 4; ++g)
#pragma unroll
        for (int e = 0; e < 8; ++e) acc[j][g][e] = bvals[j][g];

    // opaque zero offset, redefined every iteration: defeats LICM on the weight
    // loads while keeping the pointer's global address space inferable.
    unsigned zoff = 0;
    asm volatile("" : "+s"(zoff));
    const char* bp = bb0 + zoff;

    // rolling A byte pointers (64 B per K-block; lane-half folded in)
    const char* ax = (const char*)(in + ((size_t)rowA * T + tt) * Kx + 8 * half);
    const char* ah = (const char*)(hcur + (size_t)rowA * HDIM + 8 * half);

    // pipeline preamble: A of block 0, B frags 0 and 1 of block 0
    v16bf a    = load_afrag(ax);
    v16bf bcur = load_bfrag(bp, 0);
    v16bf bnx1 = load_bfrag(bp, 1);
    v16bf an;

    // ---- x-region blocks 0..KBX-2 (A prefetch from the rolling x pointer)
#pragma unroll 1
    for (int kb = 0; kb < KBX - 1; ++kb) {
      an = load_afrag(ax + 64);
      WMMA_BLOCK(bp);
      bp += 65536; ax += 64; a = an;
    }
    // ---- last x block (A prefetch switches to the h region)
    {
      an = load_afrag(ah);
      WMMA_BLOCK(bp);
      bp += 65536; a = an;
    }
    // ---- h-region blocks 0..KBH-2
#pragma unroll 1
    for (int kb = 0; kb < KBH - 1; ++kb) {
      an = load_afrag(ah + 64);
      WMMA_BLOCK(bp);
      bp += 65536; ah += 64; a = an;
    }
    // ---- final h block (lookahead B loads read the pad)
    WMMA_BLOCK(bp);

    // ---- fused gates + register-resident c update
    __hip_bfloat16* hnext = hstate + ((size_t)d * 2 + ((t + 1) & 1)) * 65536;
#pragma unroll
    for (int j = 0; j < 2; ++j) {
      int col = (cgb + j) * 16 + nl15;
#pragma unroll
      for (int v = 0; v < 8; ++v) {
        int row = m0 + v + 8 * half;
        float ig = sigm(acc[j][0][v]);
        float fg = sigm(acc[j][1][v]);
        float gt = tanhfast(acc[j][2][v]);
        float og = sigm(acc[j][3][v]);
        float cn = fg * creg[j][v] + ig * gt;
        creg[j][v] = cn;
        float hn = og * tanhfast(cn);
        size_t sidx = (size_t)row * HDIM + col;
        hnext[sidx] = __float2bfloat16(hn);
        size_t oidx = ((size_t)row * T + tt) * 512 + d * HDIM + col;
        outBf[oidx] = __float2bfloat16(hn);
        if (lastLayer) {
          float rr = resid ? resid[((size_t)row * Tp + tt / rep) * 512 + d * HDIM + col] : 0.f;
          outF32[oidx] = hn + rr;
        }
      }
    }
    grid_barrier(bar);
  }
}

// ---------------- host orchestration ----------------
extern "C" void kernel_launch(void* const* d_in, const int* in_sizes, int n_in,
                              void* d_out, int out_size, void* d_ws, size_t ws_size,
                              hipStream_t stream) {
  (void)in_sizes; (void)n_in; (void)out_size; (void)ws_size;
  const int nl[5] = {2, 2, 4, 4, 6};
  const int Ts[5] = {4, 16, 32, 96, 288};
  const int rp[5] = {1, 4, 2, 3, 3};
  int pbase[5];
  { int ix = 12; for (int s = 0; s < 5; ++s) { pbase[s] = ix; ix += nl[s] * 6; } }
  const int wbase = 120;
  static const int hoff[5] = {0, 1024, 5120, 13312, 37888};

  char* ws = (char*)d_ws;
  size_t off = 0;
  auto alloc = [&](size_t bytes) -> char* {
    off = (off + 255) & ~(size_t)255;
    char* p = ws + off; off += bytes; return p;
  };
  float* bufA = (float*)alloc((size_t)BATCH * 288 * 512 * 4);  // even stages out
  float* bufB = (float*)alloc((size_t)BATCH * 96 * 512 * 4);   // odd stages out
  __hip_bfloat16* ping0 = (__hip_bfloat16*)alloc((size_t)BATCH * 288 * 512 * 2);
  __hip_bfloat16* ping1 = (__hip_bfloat16*)alloc((size_t)BATCH * 288 * 512 * 2);
  // +128KB pad: B-fragment lookahead may read past the last block of the last layer
  __hip_bfloat16* wpack = (__hip_bfloat16*)alloc((size_t)26688 * 1024 * 2 + 131072);
  __hip_bfloat16* hstate = (__hip_bfloat16*)alloc((size_t)2 * 2 * 65536 * 2);
  unsigned* bar = (unsigned*)alloc(256);

  bar_init<<<1, 1, 0, stream>>>(bar);

  // pack all 18 layers x 2 dirs of [Wih;Whh]^T into fragment-native bf16
  size_t wofs[18][2];
  { size_t wo = 0; int li = 0;
    for (int s = 0; s < 5; ++s)
      for (int l = 0; l < nl[s]; ++l, ++li) {
        int DinReal = (s == 0 && l == 0) ? 2 : 512;
        int Kx = (s == 0 && l == 0) ? 32 : 512;
        int total = (Kx + HDIM) * 1024;
        for (int dir = 0; dir < 2; ++dir) {
          const float* Wih = (const float*)d_in[pbase[s] + l * 6 + dir * 3 + 0];
          const float* Whh = (const float*)d_in[pbase[s] + l * 6 + dir * 3 + 1];
          wofs[li][dir] = wo;
          pack_w<<<(total + 255) / 256, 256, 0, stream>>>(Wih, Whh, wpack + wo, Kx, DinReal);
          wo += (size_t)total;
        }
      }
  }

  float* outPrev = nullptr; int Tp = 1; int li = 0;
  for (int s = 0; s < 5; ++s) {
    const int T = Ts[s];
    float* outCur = (s & 1) ? bufB : bufA;

    if (s == 0)
      gather0<<<(BATCH * 4 * 32 + 255) / 256, 256, 0, stream>>>((const float*)d_in[0], ping0);
    else {
      size_t tot = (size_t)BATCH * T * 512;
      repeat_bf16<<<(int)((tot + 255) / 256), 256, 0, stream>>>(outPrev, ping0, T, rp[s], Tp);
    }

    const float* h0 = (const float*)d_in[2 + 2 * s];
    const float* c0 = (const float*)d_in[3 + 2 * s];
    __hip_bfloat16* cur = ping0;
    __hip_bfloat16* nxt = ping1;
    for (int l = 0; l < nl[s]; ++l, ++li) {
      int last = (l == nl[s] - 1) ? 1 : 0;
      const float* rs = (last && s > 0) ? outPrev : nullptr;
      const float* bF = (const float*)d_in[pbase[s] + l * 6 + 2];
      const float* bB = (const float*)d_in[pbase[s] + l * 6 + 5];
      const float* h0f = h0 + (size_t)(2 * l) * BATCH * HDIM;
      const float* c0f = c0 + (size_t)(2 * l) * BATCH * HDIM;
      const float* h0b = h0 + (size_t)(2 * l + 1) * BATCH * HDIM;
      const float* c0b = c0 + (size_t)(2 * l + 1) * BATCH * HDIM;
      if (s == 0 && l == 0)
        lstm_layer<1><<<NWG, 256, 0, stream>>>(
            cur, T, wpack + wofs[li][0], wpack + wofs[li][1], bF, bB,
            h0f, c0f, h0b, c0b, outCur, nxt, last, rs, rp[s], Tp,
            hstate, bar);
      else
        lstm_layer<16><<<NWG, 256, 0, stream>>>(
            cur, T, wpack + wofs[li][0], wpack + wofs[li][1], bF, bB,
            h0f, c0f, h0b, c0b, outCur, nxt, last, rs, rp[s], Tp,
            hstate, bar);
      __hip_bfloat16* tsw = cur; cur = nxt; nxt = tsw;
    }

    head_k<<<(BATCH * T + 255) / 256, 256, 0, stream>>>(
        outCur, (const float*)d_in[wbase + 2 * s], (const float*)d_in[wbase + 2 * s + 1],
        (float*)d_out + hoff[s], T);

    outPrev = outCur; Tp = T;
  }
}
